// ChatGLM2Attention_10505490006148
// MI455X (gfx1250) — matrix operand order repaired and checked by reading, not compiled, not run on hardware
//
#include <hip/hip_runtime.h>
#include <hip/hip_bf16.h>

// ---------------------------------------------------------------------------
// ChatGLM2 attention block for MI455X (gfx1250), wave32 + WMMA bf16.
// Pipeline: [gemm_bf16 QKV] -> [rope/split] -> [flash attention] -> [gemm_bf16 O]
// All matmuls use v_wmma_f32_16x16x32_bf16 (fp32 accumulate).
// Attention K-tile staging uses GLOBAL_LOAD_ASYNC_TO_LDS_B128 when available.
// ---------------------------------------------------------------------------

typedef __attribute__((ext_vector_type(16))) __bf16 v16bf;
typedef __attribute__((ext_vector_type(8)))  float  v8f;
typedef __attribute__((ext_vector_type(4)))  int    v4i;

typedef __attribute__((address_space(1))) v4i* gptr_v4i;   // global
typedef __attribute__((address_space(3))) v4i* lptr_v4i;   // LDS

union Frag16 {            // one 16x32 bf16 A-operand or 32x16 B-operand
    v16bf v;
    uint4 q[2];
    unsigned short u[16];
};

#if defined(__gfx1250__) && __has_builtin(__builtin_amdgcn_global_load_async_to_lds_b128)
#define ASYNC_K_STAGE 1
#else
#define ASYNC_K_STAGE 0
#endif

static __device__ inline unsigned short f2bf(float f) {
    unsigned int u = __float_as_uint(f);
    u += 0x7FFFu + ((u >> 16) & 1u);      // round-to-nearest-even
    return (unsigned short)(u >> 16);
}

static __device__ inline v8f v8f_zero() {
    v8f z;
    #pragma unroll
    for (int i = 0; i < 8; ++i) z[i] = 0.0f;
    return z;
}

static __device__ inline float half_reduce_max(float x) {
    #pragma unroll
    for (int m = 1; m < 16; m <<= 1) x = fmaxf(x, __shfl_xor(x, m, 32));
    return x;
}
static __device__ inline float half_reduce_sum(float x) {
    #pragma unroll
    for (int m = 1; m < 16; m <<= 1) x += __shfl_xor(x, m, 32);
    return x;
}

// ---------------------------------------------------------------------------
// Kernel 1/4: C[M][N] = A[M][K] * W[N][K]^T + bias   (fp32 in/out, bf16 WMMA)
// Block tile 128x128, 8 waves arranged 4(M) x 2(N): each wave 32x64 = 2x4 WMMA.
// ---------------------------------------------------------------------------
#define GK   32      // K chunk per LDS stage
#define GLD  48      // LDS row stride in halves (32 + 16 pad, keeps 16B align)

__global__ void __launch_bounds__(256)
gemm_bf16_wmma(const float* __restrict__ A,     // [M][K]
               const float* __restrict__ W,     // [N][K]
               const float* __restrict__ bias,  // [N] or nullptr
               float* __restrict__ C,           // [M][N]
               int M, int N, int K)
{
    __shared__ __align__(16) unsigned short As[128 * GLD];
    __shared__ __align__(16) unsigned short Bs[128 * GLD];

    const int tid  = threadIdx.x;
    const int lane = tid & 31;
    const int wave = tid >> 5;
    const int l16  = lane & 15;
    const int h    = lane >> 4;          // half-wave select
    const int bm   = blockIdx.x * 128;
    const int bn   = blockIdx.y * 128;
    const int wm   = (wave & 3) * 32;    // wave's M offset in tile
    const int wn   = (wave >> 2) * 64;   // wave's N offset in tile

    v8f acc[2][4];
    #pragma unroll
    for (int i = 0; i < 2; ++i)
        #pragma unroll
        for (int j = 0; j < 4; ++j) acc[i][j] = v8f_zero();

    const int lrow = tid >> 3;           // 0..31
    const int lcol = (tid & 7) * 4;      // 0..28

    for (int k0 = 0; k0 < K; k0 += GK) {
        // ---- stage A/B 128x32 fp32 -> bf16 LDS ----
        #pragma unroll
        for (int p = 0; p < 4; ++p) {
            const int r = lrow + p * 32;
            float4 a4 = *(const float4*)&A[(size_t)(bm + r) * K + k0 + lcol];
            float4 b4 = *(const float4*)&W[(size_t)(bn + r) * K + k0 + lcol];
            *(uint2*)&As[r * GLD + lcol] =
                make_uint2((unsigned)f2bf(a4.x) | ((unsigned)f2bf(a4.y) << 16),
                           (unsigned)f2bf(a4.z) | ((unsigned)f2bf(a4.w) << 16));
            *(uint2*)&Bs[r * GLD + lcol] =
                make_uint2((unsigned)f2bf(b4.x) | ((unsigned)f2bf(b4.y) << 16),
                           (unsigned)f2bf(b4.z) | ((unsigned)f2bf(b4.w) << 16));
        }
        // prefetch next K-chunk while this one computes (global_prefetch_b8)
        if (k0 + GK < K) {
            __builtin_prefetch(&A[(size_t)(bm + lrow) * K + k0 + GK + lcol], 0, 3);
            __builtin_prefetch(&W[(size_t)(bn + lrow) * K + k0 + GK + lcol], 0, 3);
        }
        __syncthreads();

        // ---- fragments ----
        Frag16 af[2], bfr[4];
        #pragma unroll
        for (int mt = 0; mt < 2; ++mt) {
            const int row = wm + mt * 16 + l16;            // A: lane = row
            af[mt].q[0] = *(const uint4*)&As[row * GLD + 8 * h];
            af[mt].q[1] = *(const uint4*)&As[row * GLD + 16 + 8 * h];
        }
        #pragma unroll
        for (int nt = 0; nt < 4; ++nt) {
            const int row = wn + nt * 16 + l16;            // B: lane = column
            bfr[nt].q[0] = *(const uint4*)&Bs[row * GLD + 8 * h];
            bfr[nt].q[1] = *(const uint4*)&Bs[row * GLD + 16 + 8 * h];
        }
        #pragma unroll
        for (int mt = 0; mt < 2; ++mt)
            #pragma unroll
            for (int nt = 0; nt < 4; ++nt)
                acc[mt][nt] = __builtin_amdgcn_wmma_f32_16x16x32_bf16(
                    false, af[mt].v, false, bfr[nt].v,
                    (short)0, acc[mt][nt], false, false);
        __syncthreads();
    }

    // ---- epilogue: C layout = VGPR r -> row (r + 8h), lane&15 -> col ----
    #pragma unroll
    for (int mt = 0; mt < 2; ++mt) {
        const int mbase = bm + wm + mt * 16 + 8 * h;
        #pragma unroll
        for (int nt = 0; nt < 4; ++nt) {
            const int col = bn + wn + nt * 16 + l16;
            const float bv = bias ? bias[col] : 0.0f;
            #pragma unroll
            for (int r = 0; r < 8; ++r)
                C[(size_t)(mbase + r) * N + col] = acc[mt][nt][r] + bv;
        }
    }
}

// ---------------------------------------------------------------------------
// Kernel 2/4: RoPE + QKV split.  fused [S][4608] fp32 ->
//   qb [S][32][128] bf16 (scaled by 1/sqrt(D)), kb/vb [G][S][128] bf16,
//   k_cache/v_cache [S][G][128] fp32 (into d_out tail).
// ---------------------------------------------------------------------------
__global__ void __launch_bounds__(256)
rope_split_kernel(const float* __restrict__ fused,
                  const float* __restrict__ rope,   // [S][64][2]
                  unsigned short* __restrict__ qb,
                  unsigned short* __restrict__ kb,
                  unsigned short* __restrict__ vb,
                  float* __restrict__ kcache,
                  float* __restrict__ vcache)
{
    const int s = blockIdx.x;
    const float* f  = fused + (size_t)s * 4608;
    const float* rp = rope  + (size_t)s * 128;
    const float scale = 0.08838834764831845f;   // 1/sqrt(128)

    // q: 32 heads * 64 rotary pairs
    for (int j = threadIdx.x; j < 2048; j += 256) {
        const int hd = j >> 6, pj = j & 63;
        const float tr = f[hd * 128 + 2 * pj], ti = f[hd * 128 + 2 * pj + 1];
        const float fr = rp[2 * pj], fi = rp[2 * pj + 1];
        const size_t base = ((size_t)s * 32 + hd) * 128 + 2 * pj;
        qb[base]     = f2bf((tr * fr - ti * fi) * scale);
        qb[base + 1] = f2bf((tr * fi + ti * fr) * scale);
    }
    // k: 2 groups * 64 pairs (+ fp32 cache)
    for (int j = threadIdx.x; j < 128; j += 256) {
        const int g = j >> 6, pj = j & 63;
        const float tr = f[4096 + g * 128 + 2 * pj], ti = f[4096 + g * 128 + 2 * pj + 1];
        const float fr = rp[2 * pj], fi = rp[2 * pj + 1];
        const float orr = tr * fr - ti * fi, oi = tr * fi + ti * fr;
        const size_t kbase = ((size_t)g * 2048 + s) * 128 + 2 * pj;
        kb[kbase] = f2bf(orr);  kb[kbase + 1] = f2bf(oi);
        const size_t cbase = ((size_t)s * 2 + g) * 128 + 2 * pj;
        kcache[cbase] = orr;    kcache[cbase + 1] = oi;
    }
    // v: 2 groups * 128 elems (+ fp32 cache)
    for (int j = threadIdx.x; j < 256; j += 256) {
        const int g = j >> 7, d = j & 127;
        const float vv = f[4352 + g * 128 + d];
        vb[((size_t)g * 2048 + s) * 128 + d] = f2bf(vv);
        vcache[((size_t)s * 2 + g) * 128 + d] = vv;
    }
}

// ---------------------------------------------------------------------------
// Kernel 3/4: flash attention, causal, GQA (16 q-heads per kv group).
// Grid (16 m-tiles, 32 heads); 8 waves, each owns 16 query rows.
// Per 64-key block: 16 WMMA for QK^T, online softmax, 16 WMMA for P*V.
// ---------------------------------------------------------------------------
#define KLD 144   // K tile row stride (128 + 16 pad), halves
#define VLD 80    // V^T row stride (64 + 16 pad), halves
#define PLD 64    // P scratch row stride, halves

__global__ void __launch_bounds__(256)
attn_kernel(const unsigned short* __restrict__ qb,  // [S][32][128] bf16, pre-scaled
            const unsigned short* __restrict__ kb,  // [G][S][128]  bf16
            const unsigned short* __restrict__ vb,  // [G][S][128]  bf16
            float* __restrict__ attn)               // [S][4096] fp32
{
    __shared__ __align__(16) unsigned short Ks[64 * KLD];        // 18 KB
    __shared__ __align__(16) unsigned short Vt[128 * VLD];       // 20 KB (transposed)
    __shared__ __align__(16) unsigned short Ps[8 * 16 * PLD];    // 16 KB

    const int mtile = blockIdx.x;          // 0..15 (128 query rows each)
    const int head  = blockIdx.y;          // 0..31
    const int g     = head >> 4;

    const int tid  = threadIdx.x;
    const int lane = tid & 31;
    const int wave = tid >> 5;
    const int l16  = lane & 15;
    const int h    = lane >> 4;
    const int row0 = mtile * 128 + wave * 16;   // wave's first query row

    // Q fragments (16 rows x 128 d): 4 chunks of K=32, resident in VGPRs
    Frag16 qf[4];
    {
        const unsigned short* qrow = qb + ((size_t)(row0 + l16) * 32 + head) * 128;
        #pragma unroll
        for (int kc = 0; kc < 4; ++kc) {
            qf[kc].q[0] = *(const uint4*)&qrow[kc * 32 + 8 * h];
            qf[kc].q[1] = *(const uint4*)&qrow[kc * 32 + 16 + 8 * h];
        }
    }

    v8f o[8];
    #pragma unroll
    for (int i = 0; i < 8; ++i) o[i] = v8f_zero();
    float mstate[8], lstate[8];
    #pragma unroll
    for (int r = 0; r < 8; ++r) { mstate[r] = -1e30f; lstate[r] = 0.0f; }

    const int nkb = 2 * mtile + 2;           // causal: key blocks 0..2m+1
    for (int kbi = 0; kbi < nkb; ++kbi) {
        const int t0 = kbi * 64;

        // ---- stage K (row-major, async when available) and V (transposed) ----
        {
            const int r = tid >> 2;          // 0..63
            const int c = (tid & 3) * 32;    // 0,32,64,96
            unsigned short* ksrc = const_cast<unsigned short*>(
                kb + ((size_t)g * 2048 + t0 + r) * 128 + c);
            const unsigned short* vsrc = vb + ((size_t)g * 2048 + t0 + r) * 128 + c;
#if ASYNC_K_STAGE
            #pragma unroll
            for (int i = 0; i < 4; ++i)
                __builtin_amdgcn_global_load_async_to_lds_b128(
                    (gptr_v4i)(v4i*)(ksrc + 8 * i),
                    (lptr_v4i)(v4i*)&Ks[r * KLD + c + 8 * i],
                    0, 0);
#else
            #pragma unroll
            for (int i = 0; i < 4; ++i)
                *(uint4*)&Ks[r * KLD + c + 8 * i] = *(const uint4*)&ksrc[8 * i];
#endif
            #pragma unroll
            for (int i = 0; i < 32; ++i)
                Vt[(c + i) * VLD + r] = vsrc[i];
#if ASYNC_K_STAGE
#if __has_builtin(__builtin_amdgcn_s_wait_asynccnt)
            __builtin_amdgcn_s_wait_asynccnt(0);
#else
            asm volatile("s_wait_asynccnt 0x0" ::: "memory");
#endif
#endif
        }
        __syncthreads();

        // ---- S = Q * K^T  (4 key n-tiles x 4 d-chunks) ----
        v8f sc[4];
        #pragma unroll
        for (int j = 0; j < 4; ++j) {
            v8f s = v8f_zero();
            const int trow = j * 16 + l16;   // B: lane = key column
            #pragma unroll
            for (int kc = 0; kc < 4; ++kc) {
                Frag16 kf;
                kf.q[0] = *(const uint4*)&Ks[trow * KLD + kc * 32 + 8 * h];
                kf.q[1] = *(const uint4*)&Ks[trow * KLD + kc * 32 + 16 + 8 * h];
                s = __builtin_amdgcn_wmma_f32_16x16x32_bf16(
                        false, qf[kc].v, false, kf.v, (short)0, s, false, false);
            }
            sc[j] = s;
        }

        // ---- causal mask + online softmax (rows = r + 8h per C layout) ----
        float bmax[8];
        #pragma unroll
        for (int r = 0; r < 8; ++r) bmax[r] = -1e30f;
        #pragma unroll
        for (int j = 0; j < 4; ++j) {
            const int tcol = t0 + j * 16 + l16;
            #pragma unroll
            for (int r = 0; r < 8; ++r) {
                const int srow = row0 + r + 8 * h;
                float v = sc[j][r] + ((tcol > srow) ? -1e9f : 0.0f);
                sc[j][r] = v;
                bmax[r] = fmaxf(bmax[r], v);
            }
        }
        float alpha[8];
        #pragma unroll
        for (int r = 0; r < 8; ++r) {
            const float bmx = half_reduce_max(bmax[r]);
            const float mn = fmaxf(mstate[r], bmx);
            alpha[r] = __expf(mstate[r] - mn);
            mstate[r] = mn;
        }
        float rsum[8];
        #pragma unroll
        for (int r = 0; r < 8; ++r) rsum[r] = 0.0f;
        #pragma unroll
        for (int j = 0; j < 4; ++j)
            #pragma unroll
            for (int r = 0; r < 8; ++r) {
                const float p = __expf(sc[j][r] - mstate[r]);
                rsum[r] += p;
                Ps[(wave * 16 + r + 8 * h) * PLD + j * 16 + l16] = f2bf(p);
            }
        #pragma unroll
        for (int r = 0; r < 8; ++r) {
            lstate[r] = lstate[r] * alpha[r] + half_reduce_sum(rsum[r]);
            #pragma unroll
            for (int dt = 0; dt < 8; ++dt) o[dt][r] *= alpha[r];
        }

        // ---- O += P * V  (2 key chunks of 32 x 8 d-tiles) ----
        const unsigned short* prow = &Ps[(wave * 16 + l16) * PLD];  // A: lane = row
        #pragma unroll
        for (int tc = 0; tc < 2; ++tc) {
            Frag16 pf;
            pf.q[0] = *(const uint4*)&prow[tc * 32 + 8 * h];
            pf.q[1] = *(const uint4*)&prow[tc * 32 + 16 + 8 * h];
            #pragma unroll
            for (int dt = 0; dt < 8; ++dt) {
                Frag16 vf;
                const int drow = dt * 16 + l16;                     // B: lane = d column
                vf.q[0] = *(const uint4*)&Vt[drow * VLD + tc * 32 + 8 * h];
                vf.q[1] = *(const uint4*)&Vt[drow * VLD + tc * 32 + 16 + 8 * h];
                o[dt] = __builtin_amdgcn_wmma_f32_16x16x32_bf16(
                            false, pf.v, false, vf.v, (short)0, o[dt], false, false);
            }
        }
        __syncthreads();   // protect Ks/Vt before next stage
    }

    // ---- normalize + store [s][head*128 + d] ----
    #pragma unroll
    for (int dt = 0; dt < 8; ++dt) {
        const int d = dt * 16 + l16;
        #pragma unroll
        for (int r = 0; r < 8; ++r) {
            const int srow = row0 + r + 8 * h;
            attn[(size_t)srow * 4096 + head * 128 + d] = o[dt][r] / lstate[r];
        }
    }
}

// ---------------------------------------------------------------------------
// Launch.  d_in: x, rotary_emb, attention_mask(unused), Wqkv, bqkv, Wo.
// d_out: out (2048*4096) | k_cache (2048*2*128) | v_cache (2048*2*128), fp32.
// Workspace layout (needs ~91 MB):
//   fused fp32 [2048][4608] | attn fp32 [2048][4096] | qb bf16 | kb bf16 | vb bf16
// ---------------------------------------------------------------------------
extern "C" void kernel_launch(void* const* d_in, const int* in_sizes, int n_in,
                              void* d_out, int out_size, void* d_ws, size_t ws_size,
                              hipStream_t stream) {
    (void)in_sizes; (void)n_in; (void)out_size; (void)ws_size;
    const float* x    = (const float*)d_in[0];
    const float* rope = (const float*)d_in[1];
    const float* Wqkv = (const float*)d_in[3];
    const float* bqkv = (const float*)d_in[4];
    const float* Wo   = (const float*)d_in[5];

    float* out    = (float*)d_out;
    float* kcache = out + (size_t)2048 * 4096;
    float* vcache = kcache + (size_t)2048 * 2 * 128;

    float* fused = (float*)d_ws;                                   // 9,437,184 f32
    float* attn  = fused + (size_t)2048 * 4608;                    // 8,388,608 f32
    unsigned short* qb = (unsigned short*)(attn + (size_t)2048 * 4096);
    unsigned short* kb = qb + (size_t)2048 * 4096;
    unsigned short* vb = kb + (size_t)2048 * 256;

    // 1) fused = x @ Wqkv^T + bqkv
    gemm_bf16_wmma<<<dim3(16, 36), 256, 0, stream>>>(x, Wqkv, bqkv, fused,
                                                     2048, 4608, 4096);
    // 2) RoPE + split + caches
    rope_split_kernel<<<2048, 256, 0, stream>>>(fused, rope, qb, kb, vb,
                                                kcache, vcache);
    // 3) causal flash attention
    attn_kernel<<<dim3(16, 32), 256, 0, stream>>>(qb, kb, vb, attn);
    // 4) out = attn @ Wo^T
    gemm_bf16_wmma<<<dim3(16, 32), 256, 0, stream>>>(attn, Wo, nullptr, out,
                                                     2048, 4096, 4096);
}
